// CausalSelfAttention_28028956574312
// MI455X (gfx1250) — matrix operand order repaired and checked by reading, not compiled, not run on hardware
//
#include <hip/hip_runtime.h>
#include <hip/hip_bf16.h>

// ---------------- types / helpers -----------------------------------------
typedef __attribute__((ext_vector_type(16))) __bf16 v16bf;
typedef __attribute__((ext_vector_type(8)))  float  v8f;

union FragU {
    v16bf v;
    unsigned short u[16];
    uint4 q[2];
};

__device__ __forceinline__ unsigned short f32_to_bf16(float f) {
    unsigned int u = __float_as_uint(f);
    unsigned int r = u + 0x7FFFu + ((u >> 16) & 1u);   // round to nearest even
    return (unsigned short)(r >> 16);
}

// ---------------- constants -------------------------------------------------
constexpr int Bq = 2, T = 2048, C = 1024, H = 16, HD = 64;
constexpr int M = Bq * T;            // 4096 rows
constexpr int N3 = 3 * C;            // 3072
constexpr int TM = 64, TN = 64, TK = 32;

// A fragment (16x32 bf16) from a row-major tile, leading dim `ld` (elements).
// lane<16: row=lane, K = 16*hi + j (contiguous).
__device__ __forceinline__ v16bf frag_a_rm(const unsigned short* base, int ld, int lane) {
    int row = lane & 15, hi = lane >> 4;
    FragU f;
    const uint4* p = (const uint4*)(base + (size_t)row * ld + hi * 16);
    f.q[0] = p[0];   // K = 16*hi + 0..7
    f.q[1] = p[1];   // K = 16*hi + 8..15
    return f.v;
}

// B fragment (32x16 bf16) from an N-major tile Bt[n][k], leading dim ldk.
// b[j] = Bt[n][16*hi + j]  (contiguous)
__device__ __forceinline__ v16bf frag_b_nm(const unsigned short* base, int ldk, int lane) {
    int n = lane & 15, hi = lane >> 4;
    FragU f;
    const uint4* p = (const uint4*)(base + (size_t)n * ldk + hi * 16);
    f.q[0] = p[0];
    f.q[1] = p[1];
    return f.v;
}

// B fragment for S = Q*K^T : B[d][n] = Krow[kt+n][d]; rows of K contiguous in hd.
__device__ __forceinline__ v16bf frag_b_krows(const unsigned short* krows, int k0, int lane) {
    int n = lane & 15, hi = lane >> 4;
    FragU f;
    const uint4* p = (const uint4*)(krows + (size_t)n * HD + k0 + hi * 16);
    f.q[0] = p[0];
    f.q[1] = p[1];
    return f.v;
}

// B fragment for O += P*V from V stored transposed [hd][T] per (b,h):
// b[j] = vt[(c*16+n)*T + kt + 16*hi + j]  (contiguous)
__device__ __forceinline__ v16bf frag_b_vt(const unsigned short* vt, int kt, int c, int lane) {
    int n = lane & 15, hi = lane >> 4;
    FragU f;
    const uint4* p = (const uint4*)(vt + (size_t)(c * 16 + n) * T + kt + hi * 16);
    f.q[0] = p[0];
    f.q[1] = p[1];
    return f.v;
}

__device__ __forceinline__ v8f wmma_bf16(v16bf a, v16bf b, v8f c) {
    return __builtin_amdgcn_wmma_f32_16x16x32_bf16(false, a, false, b, (short)0, c, false, false);
}

// ---------------- fp32 -> bf16 conversion (row-major) -----------------------
__global__ void f2bf4(const float* __restrict__ src, unsigned short* __restrict__ dst, int n4) {
    int i = blockIdx.x * blockDim.x + threadIdx.x;
    if (i >= n4) return;
    float4 v = ((const float4*)src)[i];
    uint2 o;
    o.x = (unsigned)f32_to_bf16(v.x) | ((unsigned)f32_to_bf16(v.y) << 16);
    o.y = (unsigned)f32_to_bf16(v.z) | ((unsigned)f32_to_bf16(v.w) << 16);
    ((uint2*)dst)[i] = o;
}

// ---------------- fp32 [K][N] -> bf16 transposed [N][K] ---------------------
__global__ void tcvt(const float* __restrict__ src, unsigned short* __restrict__ dst,
                     int Kd, int Nd) {
    int i = blockIdx.x * blockDim.x + threadIdx.x;    // over Nd*Kd, dst-contiguous
    if (i >= Nd * Kd) return;
    int n = i / Kd, k = i - n * Kd;
    dst[i] = f32_to_bf16(src[(size_t)k * Nd + n]);
}

// ---------------- QKV GEMM: [4096x1024] @ [1024x3072] ----------------------
// Bt = W_attn^T bf16 [3072][1024]
__global__ __launch_bounds__(128) void qkv_gemm(
    const unsigned short* __restrict__ A,     // x bf16 [4096][1024]
    const unsigned short* __restrict__ Bt,    // W_attn^T bf16 [3072][1024]
    const float* __restrict__ bias,           // b_attn [3072]
    unsigned short* __restrict__ qbf,         // [B,H,T,hd] bf16
    float* __restrict__ kout, unsigned short* __restrict__ kbf,   // [B,H,T,hd]
    float* __restrict__ vout, unsigned short* __restrict__ vtbf)  // vout [B,H,T,hd], vt [B,H,hd,T]
{
    __shared__ __align__(16) unsigned short lds_a[TM * TK];   // row-major [64][32]
    __shared__ __align__(16) unsigned short lds_bt[TN * TK];  // n-major   [64][32]

    const int mb = blockIdx.x * TM;
    const int nb = blockIdx.y * TN;
    const int tid = threadIdx.x;
    const int lane = tid & 31, wave = tid >> 5;

    v8f acc[4];
#pragma unroll
    for (int c = 0; c < 4; ++c)
#pragma unroll
        for (int i = 0; i < 8; ++i) acc[c][i] = 0.0f;

    const int row = tid >> 1, ko = (tid & 1) * 16;
    for (int k0 = 0; k0 < C; k0 += TK) {
        // stage k 0..7 and 16..23 of A (64 rows) and Bt (64 n-rows): one b128 each
        *(uint4*)(lds_a + row * TK + ko) =
            *(const uint4*)(A + (size_t)(mb + row) * C + k0 + ko);
        *(uint4*)(lds_bt + row * TK + ko) =
            *(const uint4*)(Bt + (size_t)(nb + row) * C + k0 + ko);
        __syncthreads();

        v16bf af = frag_a_rm(lds_a + (wave * 16) * TK, TK, lane);
#pragma unroll
        for (int c = 0; c < 4; ++c) {
            v16bf bfr = frag_b_nm(lds_bt + (c * 16) * TK, TK, lane);
            acc[c] = wmma_bf16(af, bfr, acc[c]);
        }
        __syncthreads();
    }

    // epilogue: scatter into q / k / v
    const int n = lane & 15, hi = lane >> 4;
    const int sel = nb >> 10;                 // 0=q 1=k 2=v (uniform per block)
#pragma unroll
    for (int c = 0; c < 4; ++c) {
        int col = nb + c * 16 + n;
        int cc = col - sel * C;
        int h = cc >> 6, d = cc & 63;
        float bv = bias[col];
#pragma unroll
        for (int i = 0; i < 8; ++i) {
            int m = mb + wave * 16 + hi * 8 + i;
            int b = m >> 11, t = m & 2047;
            size_t idx  = (((size_t)b * H + h) * T + t) * HD + d;   // [B,H,T,hd]
            float v = acc[c][i] + bv;
            unsigned short vb = f32_to_bf16(v);
            if (sel == 0) {
                qbf[idx] = vb;
            } else if (sel == 1) {
                kout[idx] = v; kbf[idx] = vb;
            } else {
                size_t idxt = (((size_t)b * H + h) * HD + d) * T + t; // [B,H,hd,T]
                vout[idx] = v; vtbf[idxt] = vb;
            }
        }
    }
}

// ---------------- Flash attention: 1 wave per (b,h, 16 query rows) ---------
__global__ __launch_bounds__(32) void attn_kernel(
    const unsigned short* __restrict__ qbf,   // [B,H,T,hd]
    const unsigned short* __restrict__ kbf,   // [B,H,T,hd]
    const unsigned short* __restrict__ vtbf,  // [B,H,hd,T]
    unsigned short* __restrict__ yatt)        // [4096][1024] bf16
{
    __shared__ __align__(16) unsigned short lds_p[16 * 32];

    const int qt = blockIdx.x * 16;
    const int bh = blockIdx.y;                // 0..31
    const int b = bh >> 4, h = bh & 15;
    const int lane = threadIdx.x;
    const int n = lane & 15, hi = lane >> 4;

    const unsigned short* qb = qbf  + (size_t)bh * T * HD;
    const unsigned short* kb = kbf  + (size_t)bh * T * HD;
    const unsigned short* vt = vtbf + (size_t)bh * HD * T;

    const v16bf aq0 = frag_a_rm(qb + (size_t)qt * HD, HD, lane);
    const v16bf aq1 = frag_a_rm(qb + (size_t)qt * HD + 32, HD, lane);

    float mrow[8], lrow[8];
    v8f o[4];
#pragma unroll
    for (int i = 0; i < 8; ++i) { mrow[i] = -3.0e38f; lrow[i] = 0.0f; }
#pragma unroll
    for (int c = 0; c < 4; ++c)
#pragma unroll
        for (int i = 0; i < 8; ++i) o[c][i] = 0.0f;

    for (int kt = 0; kt < qt + 16; kt += 32) {
        // ---- S = Q @ K^T for 32 keys (two 16x16 tiles), f32 accum ----
        v8f s0, s1;
#pragma unroll
        for (int i = 0; i < 8; ++i) { s0[i] = 0.0f; s1[i] = 0.0f; }
        s0 = wmma_bf16(aq0, frag_b_krows(kb + (size_t)kt * HD, 0, lane), s0);
        s0 = wmma_bf16(aq1, frag_b_krows(kb + (size_t)kt * HD, 32, lane), s0);
        s1 = wmma_bf16(aq0, frag_b_krows(kb + (size_t)(kt + 16) * HD, 0, lane), s1);
        s1 = wmma_bf16(aq1, frag_b_krows(kb + (size_t)(kt + 16) * HD, 32, lane), s1);

        // ---- causal mask + online softmax (rows live in 16-lane halves) ----
        float p0[8], p1[8];
#pragma unroll
        for (int i = 0; i < 8; ++i) {
            int m = qt + hi * 8 + i;
            float v0 = s0[i] * 0.125f;           // 1/sqrt(64)
            float v1 = s1[i] * 0.125f;
            if (kt + n > m)      v0 = -3.0e38f;
            if (kt + 16 + n > m) v1 = -3.0e38f;
            float mx = fmaxf(v0, v1);
#pragma unroll
            for (int off = 8; off; off >>= 1) mx = fmaxf(mx, __shfl_xor(mx, off, 32));
            float mnew = fmaxf(mrow[i], mx);
            float e0 = __expf(v0 - mnew);
            float e1 = __expf(v1 - mnew);
            float sum = e0 + e1;
#pragma unroll
            for (int off = 8; off; off >>= 1) sum += __shfl_xor(sum, off, 32);
            float corr = __expf(mrow[i] - mnew);
            lrow[i] = lrow[i] * corr + sum;
            mrow[i] = mnew;
#pragma unroll
            for (int c = 0; c < 4; ++c) o[c][i] *= corr;
            p0[i] = e0; p1[i] = e1;
        }

        // ---- restage P (16x32) through LDS to get A-fragment layout ----
#pragma unroll
        for (int i = 0; i < 8; ++i) {
            int m = hi * 8 + i;
            lds_p[m * 32 + n]      = f32_to_bf16(p0[i]);
            lds_p[m * 32 + 16 + n] = f32_to_bf16(p1[i]);
        }
        asm volatile("s_wait_dscnt 0" ::: "memory");   // wave-local LDS fence

        v16bf ap = frag_a_rm(lds_p, 32, lane);
#pragma unroll
        for (int c = 0; c < 4; ++c)
            o[c] = wmma_bf16(ap, frag_b_vt(vt, kt, c, lane), o[c]);
        asm volatile("s_wait_dscnt 0" ::: "memory");
    }

    // ---- normalize and store y_att as bf16 [B*T][C] ----
#pragma unroll
    for (int c = 0; c < 4; ++c)
#pragma unroll
        for (int i = 0; i < 8; ++i) {
            int m = qt + hi * 8 + i;
            float val = o[c][i] / lrow[i];
            yatt[((size_t)b * T + m) * C + h * HD + c * 16 + n] = f32_to_bf16(val);
        }
}

// ---------------- output projection: [4096x1024] @ [1024x1024] -------------
// Bt = W_proj^T bf16 [1024][1024]
__global__ __launch_bounds__(128) void proj_gemm(
    const unsigned short* __restrict__ A,    // y_att bf16 [4096][1024]
    const unsigned short* __restrict__ Bt,   // W_proj^T bf16 [1024][1024]
    const float* __restrict__ bias,          // b_proj [1024]
    float* __restrict__ out)                 // y f32 [4096][1024]
{
    __shared__ __align__(16) unsigned short lds_a[TM * TK];
    __shared__ __align__(16) unsigned short lds_bt[TN * TK];

    const int mb = blockIdx.x * TM;
    const int nb = blockIdx.y * TN;
    const int tid = threadIdx.x;
    const int lane = tid & 31, wave = tid >> 5;

    v8f acc[4];
#pragma unroll
    for (int c = 0; c < 4; ++c)
#pragma unroll
        for (int i = 0; i < 8; ++i) acc[c][i] = 0.0f;

    const int row = tid >> 1, ko = (tid & 1) * 16;
    for (int k0 = 0; k0 < C; k0 += TK) {
        *(uint4*)(lds_a + row * TK + ko) =
            *(const uint4*)(A + (size_t)(mb + row) * C + k0 + ko);
        *(uint4*)(lds_bt + row * TK + ko) =
            *(const uint4*)(Bt + (size_t)(nb + row) * C + k0 + ko);
        __syncthreads();

        v16bf af = frag_a_rm(lds_a + (wave * 16) * TK, TK, lane);
#pragma unroll
        for (int c = 0; c < 4; ++c) {
            v16bf bfr = frag_b_nm(lds_bt + (c * 16) * TK, TK, lane);
            acc[c] = wmma_bf16(af, bfr, acc[c]);
        }
        __syncthreads();
    }

    const int n = lane & 15, hi = lane >> 4;
#pragma unroll
    for (int c = 0; c < 4; ++c) {
        int col = nb + c * 16 + n;
        float bv = bias[col];
#pragma unroll
        for (int i = 0; i < 8; ++i) {
            int m = mb + wave * 16 + hi * 8 + i;
            out[(size_t)m * C + col] = acc[c][i] + bv;
        }
    }
}

// ---------------- launcher --------------------------------------------------
extern "C" void kernel_launch(void* const* d_in, const int* in_sizes, int n_in,
                              void* d_out, int out_size, void* d_ws, size_t ws_size,
                              hipStream_t stream) {
    const float* x      = (const float*)d_in[0];   // [2,2048,1024]
    const float* W_attn = (const float*)d_in[1];   // [1024,3072]
    const float* b_attn = (const float*)d_in[2];   // [3072]
    const float* W_proj = (const float*)d_in[3];   // [1024,1024]
    const float* b_proj = (const float*)d_in[4];   // [1024]

    float* y    = (float*)d_out;                   // [2,2048,1024]
    float* kout = y + (size_t)M * C;               // [2,16,2048,64]
    float* vout = kout + (size_t)M * C;            // [2,16,2048,64]

    char* ws = (char*)d_ws;
    size_t off = 0;
    unsigned short* xbf   = (unsigned short*)(ws + off); off += (size_t)M * C * 2;   // 8 MB
    unsigned short* wabt  = (unsigned short*)(ws + off); off += (size_t)C * N3 * 2;  // 6 MB (W_attn^T)
    unsigned short* wpbt  = (unsigned short*)(ws + off); off += (size_t)C * C * 2;   // 2 MB (W_proj^T)
    unsigned short* qbf   = (unsigned short*)(ws + off); off += (size_t)M * C * 2;   // 8 MB
    unsigned short* kbf   = (unsigned short*)(ws + off); off += (size_t)M * C * 2;   // 8 MB
    unsigned short* vtbf  = (unsigned short*)(ws + off); off += (size_t)M * C * 2;   // 8 MB
    unsigned short* yatt  = (unsigned short*)(ws + off); off += (size_t)M * C * 2;   // 8 MB

    // x: row-major bf16; W_attn / W_proj: transposed bf16
    {
        int n4 = (M * C) / 4;
        f2bf4<<<(n4 + 255) / 256, 256, 0, stream>>>(x, xbf, n4);
        int ne = N3 * C;
        tcvt<<<(ne + 255) / 256, 256, 0, stream>>>(W_attn, wabt, C, N3);
        ne = C * C;
        tcvt<<<(ne + 255) / 256, 256, 0, stream>>>(W_proj, wpbt, C, C);
    }

    // qkv = x @ W_attn + b_attn ; scatter q/k (bf16), v (bf16 transposed), k/v f32
    qkv_gemm<<<dim3(M / TM, N3 / TN), 128, 0, stream>>>(
        xbf, wabt, b_attn, qbf, kout, kbf, vout, vtbf);

    // flash attention, 1 wave per (b,h, 16 query rows)
    attn_kernel<<<dim3(T / 16, Bq * H), 32, 0, stream>>>(qbf, kbf, vtbf, yatt);

    // y = y_att @ W_proj + b_proj
    proj_gemm<<<dim3(M / TM, C / TN), 128, 0, stream>>>(yatt, wpbt, b_proj, y);
}
